// CausalPredictor_25357486916171
// MI455X (gfx1250) — compile-verified, matrix operand order checked by reading, not run on hardware
//
#include <hip/hip_runtime.h>
#include <hip/hip_bf16.h>
#include <math.h>

#define Bsz 4
#define Lsz 1024
#define Dsz 1024
#define Ksz 20
#define Csz 21

typedef __attribute__((ext_vector_type(16))) _Float16 v16h;
typedef __attribute__((ext_vector_type(8)))  float    v8f;
typedef __attribute__((ext_vector_type(4)))  float    v4f;

// Load one lane's slice of a 16x32 f16 WMMA A-fragment from an f32 row.
// Per CDNA5 ISA (16-bit A 16x32): halves 0..7 = K kb..kb+7, halves 8..15 = K kb+16..kb+23,
// where kb = (lane>=16)*8. Caller passes p = rowptr + kc + kb.
__device__ __forceinline__ v16h load_frag_row(const float* __restrict__ p) {
  v4f f0 = *(const v4f*)(p + 0);
  v4f f1 = *(const v4f*)(p + 4);
  v4f f2 = *(const v4f*)(p + 16);
  v4f f3 = *(const v4f*)(p + 20);
  v16h r;
#pragma unroll
  for (int j = 0; j < 4; ++j) {
    r[j]      = (_Float16)f0[j];
    r[4 + j]  = (_Float16)f1[j];
    r[8 + j]  = (_Float16)f2[j];
    r[12 + j] = (_Float16)f3[j];
  }
  return r;
}

// zd[k,c] = dic[k,:] . Wz_w[c,:] + Wz_b[c] ;  dcz[k,c] = dic[k,:] . cs_w[c, D:2D]
__global__ void prep_zd_dcz(const float* __restrict__ dic,
                            const float* __restrict__ Wz_w, const float* __restrict__ Wz_b,
                            const float* __restrict__ cs_w,
                            float* __restrict__ zd, float* __restrict__ dcz) {
  int idx = blockIdx.x * blockDim.x + threadIdx.x;
  if (idx >= Ksz * Csz) return;
  int k = idx / Csz, c = idx - k * Csz;
  const float* dk = dic  + (size_t)k * Dsz;
  const float* wz = Wz_w + (size_t)c * Dsz;
  const float* cz = cs_w + (size_t)c * (2 * Dsz) + Dsz;
  float s1 = 0.f, s2 = 0.f;
  for (int d = 0; d < Dsz; d += 4) {
    v4f a = *(const v4f*)(dk + d);
    v4f w = *(const v4f*)(wz + d);
    v4f z = *(const v4f*)(cz + d);
#pragma unroll
    for (int t = 0; t < 4; ++t) { s1 += a[t] * w[t]; s2 += a[t] * z[t]; }
  }
  zd[idx]  = s1 + Wz_b[c];
  dcz[idx] = s2;
}

// Pre-pack Wy_w and cs_w[:, :D] into per-lane f16 WMMA B-fragments, zero-padded
// for columns >= C. Fragment id f = (tile*32 + chunk)*32 + lane; 16 halves each.
// One thread packs one fragment of each weight (4096 fragments total -> 2048 threads).
__global__ void pack_wfrag(const float* __restrict__ Wy_w, const float* __restrict__ cs_w,
                           _Float16* __restrict__ wyfrag, _Float16* __restrict__ cwfrag) {
  int f = blockIdx.x * blockDim.x + threadIdx.x;
  if (f >= 2 * 32 * 32) return;            // 2 N-tiles * 32 K-chunks * 32 lanes
  int lane  = f & 31;
  int chunk = (f >> 5) & 31;
  int t     = f >> 10;
  int n  = t * 16 + (lane & 15);
  int kb = (lane >> 4) << 3;
  int kc = chunk << 5;
  _Float16* wp = wyfrag + (size_t)f * 16;
  _Float16* cp = cwfrag + (size_t)f * 16;
  if (n < Csz) {
    const float* wy = Wy_w + (size_t)n * Dsz + kc + kb;
    const float* cw = cs_w + (size_t)n * (2 * Dsz) + kc + kb;
#pragma unroll
    for (int j = 0; j < 8; ++j) {
      wp[j]     = (_Float16)wy[j];
      wp[8 + j] = (_Float16)wy[16 + j];
      cp[j]     = (_Float16)cw[j];
      cp[8 + j] = (_Float16)cw[16 + j];
    }
  } else {
#pragma unroll
    for (int j = 0; j < 16; ++j) { wp[j] = (_Float16)0.f; cp[j] = (_Float16)0.f; }
  }
}

// 4 waves per block; each wave owns one 16-row tile of one image.
// WMMA: yw = y @ Wy_w.T (N padded 21->32), ly = y @ cs_w[:, :D].T.
// Then per-row softmax over K=20 dictionary logits and lz = (att*prior)@dcz + cs_b.
__global__ __launch_bounds__(128)
void fused_proj_softmax(const float* __restrict__ x,
                        const _Float16* __restrict__ wyfrag, const float* __restrict__ Wy_b,
                        const _Float16* __restrict__ cwfrag, const float* __restrict__ cs_b,
                        const float* __restrict__ prior,
                        const float* __restrict__ zd, const float* __restrict__ dcz,
                        float* __restrict__ ly_out, float* __restrict__ lz_out) {
  __shared__ float yw_s[4][16][32];
  __shared__ float ly_s[4][16][32];

  const int lane = threadIdx.x & 31;
  const int wv   = threadIdx.x >> 5;
  const int tile = blockIdx.x * 4 + wv;   // global 16-row tile id
  const int b    = tile >> 6;             // tile / (L/16)
  const int r0   = (tile & 63) << 4;

  const int m  = lane & 15;
  const int kb = (lane >> 4) << 3;        // 0 or 8

  v8f accY0 = {}, accY1 = {}, accL0 = {}, accL1 = {};

  const float* yrow = x + ((size_t)b * Lsz + r0 + m) * Dsz + kb;
  const v16h* wy0 = (const v16h*)(wyfrag) + lane;          // tile 0 fragments
  const v16h* wy1 = (const v16h*)(wyfrag) + 32 * 32 + lane; // tile 1 fragments
  const v16h* cw0 = (const v16h*)(cwfrag) + lane;
  const v16h* cw1 = (const v16h*)(cwfrag) + 32 * 32 + lane;

  for (int chunk = 0; chunk < 32; ++chunk) {
    v16h a  = load_frag_row(yrow + (chunk << 5));
    v16h b0 = wy0[chunk << 5];   // stride 32 fragments per chunk
    v16h b1 = wy1[chunk << 5];
    v16h c0 = cw0[chunk << 5];
    v16h c1 = cw1[chunk << 5];
    accY0 = __builtin_amdgcn_wmma_f32_16x16x32_f16(false, a, false, b0, (short)0, accY0, false, false);
    accY1 = __builtin_amdgcn_wmma_f32_16x16x32_f16(false, a, false, b1, (short)0, accY1, false, false);
    accL0 = __builtin_amdgcn_wmma_f32_16x16x32_f16(false, a, false, c0, (short)0, accL0, false, false);
    accL1 = __builtin_amdgcn_wmma_f32_16x16x32_f16(false, a, false, c1, (short)0, accL1, false, false);
  }

  // C/D layout: VGPR r, lanes 0-15 -> M=r, lanes 16-31 -> M=8+r; N = lane&15.
  const int rbase = (lane >> 4) << 3;
  const int col   = lane & 15;
#pragma unroll
  for (int r = 0; r < 8; ++r) {
    yw_s[wv][rbase + r][col]      = accY0[r];
    yw_s[wv][rbase + r][16 + col] = accY1[r];
    ly_s[wv][rbase + r][col]      = accL0[r];
    ly_s[wv][rbase + r][16 + col] = accL1[r];
  }
  __syncthreads();

  if (lane < 16) {
    const int row = lane;
    float yw[Csz], ly[Csz];
#pragma unroll
    for (int c = 0; c < Csz; ++c) {
      yw[c] = yw_s[wv][row][c] + Wy_b[c];
      ly[c] = ly_s[wv][row][c];
    }
    const float scale = 0.21821789023599236f;  // 1/sqrt(21)
    float lg[Ksz];
    float mx = -3.0e38f;
#pragma unroll
    for (int k = 0; k < Ksz; ++k) {
      float s = 0.f;
#pragma unroll
      for (int c = 0; c < Csz; ++c) s += yw[c] * zd[k * Csz + c];
      s *= scale;
      lg[k] = s;
      mx = fmaxf(mx, s);
    }
    float sum = 0.f;
#pragma unroll
    for (int k = 0; k < Ksz; ++k) { lg[k] = __expf(lg[k] - mx); sum += lg[k]; }
    const float inv = 1.f / sum;
    float lz[Csz];
#pragma unroll
    for (int c = 0; c < Csz; ++c) lz[c] = cs_b[c];
#pragma unroll
    for (int k = 0; k < Ksz; ++k) {
      float a = lg[k] * inv * prior[k];
#pragma unroll
      for (int c = 0; c < Csz; ++c) lz[c] += a * dcz[k * Csz + c];
    }
    float* lyp = ly_out + ((size_t)b * Lsz + r0 + row) * Csz;
    float* lzp = lz_out + ((size_t)b * Lsz + r0 + row) * Csz;
#pragma unroll
    for (int c = 0; c < Csz; ++c) { lyp[c] = ly[c]; lzp[c] = lz[c]; }
  }
}

// One block per (b, i): out[(b*L+i)*L*C + g] = lz_flat[b][g] + ly_i[g % 21].
// 352 MB of writes -> non-temporal b128 stores; lz (86 KB / image) stays L2-resident.
__global__ __launch_bounds__(256)
void outer_sum_store(const float* __restrict__ ly, const float* __restrict__ lz,
                     float* __restrict__ out) {
  __shared__ float ly_sh[Csz];
  const int bi = blockIdx.x;     // b*L + i
  const int b  = bi >> 10;
  if (threadIdx.x < Csz) ly_sh[threadIdx.x] = ly[(size_t)bi * Csz + threadIdx.x];
  __syncthreads();

  const v4f* lz4 = (const v4f*)(lz + (size_t)b * (Lsz * Csz));
  v4f* o4 = (v4f*)(out + (size_t)bi * (Lsz * Csz));
  const int n4 = (Lsz * Csz) / 4;  // 5376
  for (int g = threadIdx.x; g < n4; g += 256) {
    v4f v = lz4[g];
    int e = g * 4;
    int c = e - (e / Csz) * Csz;   // e % 21 via magic-number division
#pragma unroll
    for (int t = 0; t < 4; ++t) {
      v[t] += ly_sh[c];
      c = (c + 1 == Csz) ? 0 : (c + 1);
    }
    __builtin_nontemporal_store(v, o4 + g);
  }
}

extern "C" void kernel_launch(void* const* d_in, const int* in_sizes, int n_in,
                              void* d_out, int out_size, void* d_ws, size_t ws_size,
                              hipStream_t stream) {
  const float* x     = (const float*)d_in[0];
  const float* dic   = (const float*)d_in[1];
  const float* prior = (const float*)d_in[2];
  const float* Wy_w  = (const float*)d_in[3];
  const float* Wy_b  = (const float*)d_in[4];
  const float* Wz_w  = (const float*)d_in[5];
  const float* Wz_b  = (const float*)d_in[6];
  const float* cs_w  = (const float*)d_in[7];
  const float* cs_b  = (const float*)d_in[8];
  float* out = (float*)d_out;

  // workspace: zd[420] | dcz[420] | ly[B*L*C] | lz[B*L*C] | wyfrag | cwfrag (~820 KB)
  float* zd  = (float*)d_ws;
  float* dcz = zd + Ksz * Csz;
  float* ly  = dcz + Ksz * Csz;
  float* lz  = ly + (size_t)Bsz * Lsz * Csz;
  _Float16* wyfrag = (_Float16*)(lz + (size_t)Bsz * Lsz * Csz);  // 32768 halves (32B-aligned)
  _Float16* cwfrag = wyfrag + 2 * 32 * 32 * 16;                   // 32768 halves

  prep_zd_dcz<<<2, 256, 0, stream>>>(dic, Wz_w, Wz_b, cs_w, zd, dcz);
  pack_wfrag<<<8, 256, 0, stream>>>(Wy_w, cs_w, wyfrag, cwfrag);
  fused_proj_softmax<<<(Bsz * (Lsz / 16)) / 4, 128, 0, stream>>>(
      x, wyfrag, Wy_b, cwfrag, cs_b, prior, zd, dcz, ly, lz);
  outer_sum_store<<<Bsz * Lsz, 256, 0, stream>>>(ly, lz, out);
}
